// BSplineKANLayer_45741401702444
// MI455X (gfx1250) — matrix operand order repaired and checked by reading, not compiled, not run on hardware
//
#include <hip/hip_runtime.h>
#include <hip/hip_bf16.h>

// ---------------------------------------------------------------------------
// BSpline KAN layer for MI455X (gfx1250, wave32, WMMA).
//
// y = tanh( [basis(x) | silu(x)] @ [coeffs | base_weight]^T + res_scale * x )
//   M = 16384, N = 512, K = 4096 + 512 = 4608   (~77 GFLOP)
//
// Memory floor ~73MB -> ~3us @ 23.3 TB/s; compute-bound, so run the GEMM in
// f16 WMMA (v_wmma_f32_16x16x32_f16, f32 accumulate). Basis values are in
// [0,1] and coeffs ~0.1*N(0,1): f16 + f32 accum keeps rel.err ~1e-3.
// Tile staging uses CDNA5 async global->LDS loads (ASYNCcnt) when available.
// ---------------------------------------------------------------------------

#define IN_DIM   512
#define OUT_DIM  512
#define BATCH    16384
#define NCOEF    8
#define KSPLINE  (IN_DIM * NCOEF)     // 4096
#define KTOT     (KSPLINE + IN_DIM)   // 4608
#define KSTEP    32
#define KSTEPS   (KTOT / KSTEP)       // 144
#define LDS_STRIDE 40                 // 32 halves + 8 pad -> conflict-free

typedef __attribute__((ext_vector_type(16))) _Float16 v16h;
typedef __attribute__((ext_vector_type(8)))  _Float16 v8h;
typedef __attribute__((ext_vector_type(8)))  float    v8f;

union U16h { v16h v; v8h h[2]; };
union U8h  { v8h v; _Float16 e[8]; };

// ---- CDNA5 async global->LDS path ----------------------------------------
// Probe round 2 revealed the builtin signature: arg0 is a pointer to a
// 4 x i32 GCC-vector in AS1 (printed by clang as "int __vector_size__(16)
// __device__ *"); use exactly-typed, address-space-qualified casts.
typedef int v4i32 __attribute__((vector_size(16)));
typedef __attribute__((address_space(1))) v4i32* as1_v4i32_ptr;
typedef __attribute__((address_space(3))) v4i32* as3_v4i32_ptr;

#if __has_builtin(__builtin_amdgcn_global_load_async_to_lds_b128)
#define HAVE_ASYNC_LDS 1
#define GLOBAL_ASYNC_B128(g, l)                                       \
    __builtin_amdgcn_global_load_async_to_lds_b128(                   \
        (as1_v4i32_ptr)(g), (as3_v4i32_ptr)(l), 0, 0)
#else
#define HAVE_ASYNC_LDS 0
#endif

#if __has_builtin(__builtin_amdgcn_s_wait_asynccnt)
#define WAIT_ASYNC0() __builtin_amdgcn_s_wait_asynccnt(0)
#else
#define WAIT_ASYNC0() asm volatile("s_wait_asynccnt 0x0" ::: "memory")
#endif

// ---------------------------------------------------------------------------
// Kernel 1: knots from softplus + cumsum.  grid: (512, 12) f32 in workspace.
// ---------------------------------------------------------------------------
__global__ void grid_kernel(const float* __restrict__ gsl,
                            const float* __restrict__ gstart,
                            float* __restrict__ grid) {
    int i = blockIdx.x * blockDim.x + threadIdx.x;
    if (i >= IN_DIM) return;
    float s = gstart[i];
    grid[i * 12 + 0] = s;
    float acc = 0.0f;
#pragma unroll
    for (int j = 0; j < 11; ++j) {
        acc += log1pf(__expf(gsl[i * 11 + j]));   // softplus
        grid[i * 12 + j + 1] = s + acc;
    }
}

// ---------------------------------------------------------------------------
// Kernel 2: Cox-de Boor degree-3 basis (8 values) + silu(x), packed f16 into
// A: (BATCH, KTOT) row-major.  A[b, in*8+c] = basis, A[b, 4096+in] = silu(x).
// ---------------------------------------------------------------------------
__global__ void basis_silu_kernel(const float* __restrict__ X,
                                  const float* __restrict__ grid,
                                  _Float16* __restrict__ Ah) {
    int idx = blockIdx.x * blockDim.x + threadIdx.x;   // b*512 + in
    int b  = idx >> 9;
    int in = idx & (IN_DIM - 1);
    float xv = X[idx];

    float g[12];
#pragma unroll
    for (int j = 0; j < 12; ++j) g[j] = grid[in * 12 + j];

    float bases[11];
#pragma unroll
    for (int t = 0; t < 11; ++t)
        bases[t] = (xv >= g[t] && xv < g[t + 1]) ? 1.0f : 0.0f;

    float denom[11];
#pragma unroll
    for (int t = 0; t < 11; ++t) denom[t] = g[t + 1] - g[t];

    const float eps = 1e-8f;
#pragma unroll
    for (int k = 0; k < 3; ++k) {
#pragma unroll
        for (int t = 0; t < 10 - k; ++t) {
            float l = (xv - g[t])           / (denom[t]         + eps);
            float r = (g[t + 2 + k] - xv)   / (denom[t + 1 + k] + eps);
            bases[t] = l * bases[t] + r * bases[t + 1];
        }
    }

    U8h o;
#pragma unroll
    for (int c = 0; c < 8; ++c) o.e[c] = (_Float16)bases[c];
    *(v8h*)(Ah + (size_t)b * KTOT + in * NCOEF) = o.v;   // 16B aligned store

    float s = xv / (1.0f + __expf(-xv));                 // silu
    Ah[(size_t)b * KTOT + KSPLINE + in] = (_Float16)s;
}

// ---------------------------------------------------------------------------
// Kernel 3: pack B = [coeffs | base_weight] to f16, (OUT_DIM, KTOT) row-major.
// ---------------------------------------------------------------------------
__global__ void packB_kernel(const float* __restrict__ coeffs,
                             const float* __restrict__ bw,
                             _Float16* __restrict__ Bh) {
    int e = blockIdx.x * blockDim.x + threadIdx.x;   // < 512*4608
    int n = e / KTOT;
    int k = e - n * KTOT;
    float v = (k < KSPLINE) ? coeffs[(size_t)n * KSPLINE + k]
                            : bw[(size_t)n * IN_DIM + (k - KSPLINE)];
    Bh[e] = (_Float16)v;
}

// ---------------------------------------------------------------------------
// Kernel 4: f16 WMMA GEMM.  Block tile 128x128, 8 waves, wave tile 32x64
// (2x4 WMMA accumulators).  LDS double-buffered, row stride 40 halves.
// Tiles staged with async global->LDS b128 (no VGPR round-trip) if available.
// Epilogue: tanh(acc + res_scale * x), branch-free tanh.
// ---------------------------------------------------------------------------
__device__ __forceinline__ void load_tiles(const _Float16* __restrict__ Ah,
                                           const _Float16* __restrict__ Bh,
                                           _Float16* sAbuf, _Float16* sBbuf,
                                           int rowbase, int nbase, int kt, int tid) {
#pragma unroll
    for (int s0 = 0; s0 < 2; ++s0) {
        int s = tid + s0 * 256;          // 512 slots: 128 rows x 4 quarters
        int r = s >> 2;
        int q = s & 3;                   // 8 halves (16B) per quarter
        const _Float16* ga = Ah + (size_t)(rowbase + r) * KTOT + kt + q * 8;
        const _Float16* gb = Bh + (size_t)(nbase + r) * KTOT + kt + q * 8;
        _Float16* la = sAbuf + r * LDS_STRIDE + q * 8;
        _Float16* lb = sBbuf + r * LDS_STRIDE + q * 8;
#if HAVE_ASYNC_LDS
        GLOBAL_ASYNC_B128(ga, la);       // global_load_async_to_lds_b128
        GLOBAL_ASYNC_B128(gb, lb);
#else
        *(float4*)la = *(const float4*)ga;
        *(float4*)lb = *(const float4*)gb;
#endif
    }
}

__device__ __forceinline__ float fast_tanh(float y) {
    // tanh(y) = 1 - 2/(exp(2y)+1); exact limits at +-inf, no branches.
    float e = __expf(2.0f * y);
    return 1.0f - 2.0f / (e + 1.0f);
}

__global__ __launch_bounds__(256) void gemm_kernel(const _Float16* __restrict__ Ah,
                                                   const _Float16* __restrict__ Bh,
                                                   const float* __restrict__ X,
                                                   const float* __restrict__ res_scale,
                                                   float* __restrict__ out) {
    __shared__ _Float16 sA[2][128 * LDS_STRIDE];
    __shared__ _Float16 sB[2][128 * LDS_STRIDE];

    const int tid = threadIdx.x;
    const int tm = blockIdx.x >> 2;          // 128 M-tiles
    const int tn = blockIdx.x & 3;           // 4   N-tiles
    const int rowbase = tm * 128;
    const int nbase   = tn * 128;

    const int lane = tid & 31;
    const int lm   = lane & 15;              // M/N index within 16
    const int lhi  = lane >> 4;              // half-wave select
    const int w    = tid >> 5;               // wave 0..7
    const int m0   = (w >> 1) * 32;          // wave M offset in tile
    const int n0   = (w & 1) * 64;           // wave N offset in tile

    v8f acc[2][4];
    const v8f vzero = {0.f, 0.f, 0.f, 0.f, 0.f, 0.f, 0.f, 0.f};
#pragma unroll
    for (int i = 0; i < 2; ++i)
#pragma unroll
        for (int j = 0; j < 4; ++j) acc[i][j] = vzero;

    load_tiles(Ah, Bh, sA[0], sB[0], rowbase, nbase, 0, tid);
#if HAVE_ASYNC_LDS
    WAIT_ASYNC0();
#endif
    __syncthreads();

    for (int ks = 0; ks < KSTEPS; ++ks) {
        const int buf = ks & 1;
        if (ks + 1 < KSTEPS)
            load_tiles(Ah, Bh, sA[buf ^ 1], sB[buf ^ 1],
                       rowbase, nbase, (ks + 1) * KSTEP, tid);

        const _Float16* sa = sA[buf];
        const _Float16* sb = sB[buf];

        // A fragment: 16-bit A 16x32 layout — lane<16: K{0..7,16..23},
        // lane>=16: K{8..15,24..31}; row M = lane%16.
        v16h afrag[2];
#pragma unroll
        for (int i = 0; i < 2; ++i) {
            int row = m0 + i * 16 + lm;
            U16h u;
            u.h[0] = *(const v8h*)(sa + row * LDS_STRIDE + lhi * 8);
            u.h[1] = *(const v8h*)(sa + row * LDS_STRIDE + 16 + lhi * 8);
            afrag[i] = u.v;
        }
        // B fragment: 16-bit B 32x16 layout — lane holds column n = lane%16,
        // lane<16: K{0..7,16..23}, lane>=16: K{8..15,24..31} (matches A).
        v16h bfrag[4];
#pragma unroll
        for (int j = 0; j < 4; ++j) {
            int col = n0 + j * 16 + lm;
            U16h u;
            u.h[0] = *(const v8h*)(sb + col * LDS_STRIDE + lhi * 8);
            u.h[1] = *(const v8h*)(sb + col * LDS_STRIDE + 16 + lhi * 8);
            bfrag[j] = u.v;
        }

#pragma unroll
        for (int i = 0; i < 2; ++i)
#pragma unroll
            for (int j = 0; j < 4; ++j)
                acc[i][j] = __builtin_amdgcn_wmma_f32_16x16x32_f16(
                    /*neg_a=*/false, afrag[i], /*neg_b=*/false, bfrag[j],
                    /*c_mod=*/(short)0, acc[i][j],
                    /*reuse_a=*/false, /*reuse_b=*/false);

#if HAVE_ASYNC_LDS
        WAIT_ASYNC0();                   // next-buffer tiles landed in LDS
#endif
        __syncthreads();
    }

    // Epilogue: C/D 16x16 f32 layout — VGPR r holds M = r + 8*(lane/16),
    // N = lane%16.  y = tanh(acc + res_scale * x)
    const float rs = res_scale[0];
#pragma unroll
    for (int i = 0; i < 2; ++i) {
#pragma unroll
        for (int j = 0; j < 4; ++j) {
#pragma unroll
            for (int r = 0; r < 8; ++r) {
                int row = rowbase + m0 + i * 16 + r + 8 * lhi;
                int col = nbase + n0 + j * 16 + lm;
                float y = acc[i][j][r] + rs * X[(size_t)row * OUT_DIM + col];
                out[(size_t)row * OUT_DIM + col] = fast_tanh(y);
            }
        }
    }
}

// ---------------------------------------------------------------------------
// Launch.  Workspace layout: A f16 (BATCH*KTOT) | B f16 (OUT_DIM*KTOT) |
// grid f32 (512*12).  ~156 MB total.
// ---------------------------------------------------------------------------
extern "C" void kernel_launch(void* const* d_in, const int* in_sizes, int n_in,
                              void* d_out, int out_size, void* d_ws, size_t ws_size,
                              hipStream_t stream) {
    const float* x      = (const float*)d_in[0];
    const float* coeffs = (const float*)d_in[1];
    const float* bw     = (const float*)d_in[2];
    const float* gsl    = (const float*)d_in[3];
    const float* gstart = (const float*)d_in[4];
    const float* rscale = (const float*)d_in[5];
    float* out = (float*)d_out;

    _Float16* Ah = (_Float16*)d_ws;
    _Float16* Bh = Ah + (size_t)BATCH * KTOT;
    float* grid  = (float*)(Bh + (size_t)OUT_DIM * KTOT);

    grid_kernel<<<2, 256, 0, stream>>>(gsl, gstart, grid);

    basis_silu_kernel<<<(BATCH * IN_DIM) / 256, 256, 0, stream>>>(x, grid, Ah);

    packB_kernel<<<(OUT_DIM * KTOT) / 256, 256, 0, stream>>>(coeffs, bw, Bh);

    gemm_kernel<<<(BATCH / 128) * (OUT_DIM / 128), 256, 0, stream>>>(
        Ah, Bh, x, rscale, out);
}
